// xLSTM_14516989461038
// MI455X (gfx1250) — matrix operand order repaired and checked by reading, not compiled, not run on hardware
//
#include <hip/hip_runtime.h>
#include <hip/hip_bf16.h>
#include <cstddef>

// ---------------------------------------------------------------------------
// Types for CDNA5 WMMA (wave32): bf16 A/B fragments (16 x bf16 = 8 VGPRs),
// fp32 C/D accumulator (8 VGPRs).
// ---------------------------------------------------------------------------
typedef __attribute__((ext_vector_type(16))) __bf16 v16bf;
typedef __attribute__((ext_vector_type(8)))  float  v8f;
typedef __attribute__((ext_vector_type(4)))  unsigned int u32x4;
typedef __attribute__((ext_vector_type(8)))  int i32x8;
typedef __attribute__((ext_vector_type(4)))  int i32x4;

union BF16Frag {
    v16bf bf;
    uint4 u[2];
};

static __device__ __forceinline__ unsigned short f32_to_bf16(float f) {
    unsigned int u = __float_as_uint(f);
    // round-to-nearest-even
    unsigned int r = (u + 0x7FFFu + ((u >> 16) & 1u)) >> 16;
    return (unsigned short)r;
}

static __device__ __forceinline__ v8f wmma_bf16(const BF16Frag& a, const BF16Frag& b, v8f c) {
#if defined(__HIP_DEVICE_COMPILE__)
    // D = A(16x32 bf16) * B(32x16 bf16) + C(16x16 f32)
    c = __builtin_amdgcn_wmma_f32_16x16x32_bf16(
            /*neg_a=*/false, a.bf, /*neg_b=*/false, b.bf,
            /*c_mod=*/(short)0, c, /*reuse_a=*/false, /*reuse_b=*/false);
#endif
    return c;
}

#if defined(__HIP_DEVICE_COMPILE__) && __has_builtin(__builtin_amdgcn_tensor_load_to_lds)
#define CDNA5_HAS_TDM 1
#else
#define CDNA5_HAS_TDM 0
#endif

#if CDNA5_HAS_TDM
// Issue one TDM descriptor: load a rows x cols (bf16) 2-D tile from global
// memory (row stride in elements) into LDS at byte offset lds_addr.
// D# bit layout per CDNA5 ISA ch. 8 (group0: count/lds/global/type,
// group1: data_size, tensor_dim0/1, tile_dim0/1, tensor_dim0_stride).
static __device__ __forceinline__ void tdm_load_tile_2d(
    unsigned int lds_addr, const void* gptr,
    unsigned int rows, unsigned int cols, unsigned long long stride_elts) {
    unsigned long long ga = (unsigned long long)(uintptr_t)gptr;
    u32x4 g0;
    g0[0] = 1u;                                         // count=1, user mode
    g0[1] = lds_addr;                                   // lds_addr [63:32]
    g0[2] = (unsigned int)(ga & 0xFFFFFFFFu);           // global_addr lo
    g0[3] = (unsigned int)((ga >> 32) & 0x01FFFFFFu)    // global_addr hi
            | (2u << 30);                               // type=2 (image)
    i32x8 g1;
    g1[0] = (int)(1u << 16);                            // wg_mask=0, data_size=1 (2B)
    g1[1] = (int)((cols & 0xFFFFu) << 16);              // tensor_dim0 lo16 @63:48
    g1[2] = (int)(((cols >> 16) & 0xFFFFu)              // tensor_dim0 hi16
            | ((rows & 0xFFFFu) << 16));                // tensor_dim1 lo16
    g1[3] = (int)(((rows >> 16) & 0xFFFFu)              // tensor_dim1 hi16
            | ((cols & 0xFFFFu) << 16));                // tile_dim0 @127:112
    g1[4] = (int)(rows & 0xFFFFu);                      // tile_dim1, tile_dim2=0
    g1[5] = (int)(stride_elts & 0xFFFFFFFFu);           // dim0_stride lo32
    g1[6] = (int)((stride_elts >> 32) & 0xFFFFu);       // dim0_stride hi16
    g1[7] = 0;
    i32x4 z4 = {0, 0, 0, 0};
#if __clang_major__ >= 23
    i32x8 z8 = {0, 0, 0, 0, 0, 0, 0, 0};
    __builtin_amdgcn_tensor_load_to_lds(g0, g1, z4, z4, z8, 0);
#else
    __builtin_amdgcn_tensor_load_to_lds(g0, g1, z4, z4, 0);
#endif
}
#endif

#define HDIM  1024
#define KDIM  1024
#define BATCH 16
#define SEQ   256

// ---------------------------------------------------------------------------
// fp32 -> bf16 conversion (weights), grid-stride
// ---------------------------------------------------------------------------
__global__ void cvt_bf16_kernel(const float* __restrict__ s,
                                unsigned short* __restrict__ d, size_t n) {
    size_t i = (size_t)blockIdx.x * blockDim.x + threadIdx.x;
    size_t stride = (size_t)gridDim.x * blockDim.x;
    for (; i < n; i += stride) d[i] = f32_to_bf16(s[i]);
}

// ---------------------------------------------------------------------------
// Embedding gather: x = emb[seq], write fp32 (residual path) + bf16 (GEMM path)
// ---------------------------------------------------------------------------
__global__ void embed_kernel(const int* __restrict__ seq,
                             const float* __restrict__ emb,
                             float* __restrict__ xf,
                             unsigned short* __restrict__ xb, int n) {
    int i = blockIdx.x * blockDim.x + threadIdx.x;
    if (i >= n) return;
    int token = i >> 10;            // H == E == 1024
    int e = i & (HDIM - 1);
    float v = emb[(size_t)seq[token] * HDIM + e];
    xf[i] = v;
    xb[i] = f32_to_bf16(v);
}

// ---------------------------------------------------------------------------
// Fused sLSTM cell: gates = A1 @ W1^T + A2 @ W2^T + bias  (M=16, N=4H, K=1024)
// then elementwise exp-input-gate LSTM update.
// Each wave owns one column slice n..n+15 of H and computes the matching
// 16x16 tiles of all four gates (i at n, f at n+H, z at n+2H, o at n+3H),
// so the cell update happens entirely in the WMMA accumulator registers.
// A1/A2 (16x1024 bf16 each) are staged into LDS by the Tensor Data Mover
// (wave 0 issues two descriptors, waits on TENSORcnt, then a WG barrier).
// grid: 16 blocks x 128 threads (4 waves) -> 64 waves cover H=1024.
// ---------------------------------------------------------------------------
__global__ __launch_bounds__(128)
void slstm_cell_kernel(const unsigned short* __restrict__ A1, int lda1,
                       const unsigned short* __restrict__ W1,   // [4H, K] bf16
                       const unsigned short* __restrict__ A2, int lda2,
                       const unsigned short* __restrict__ W2,   // [4H, K] bf16
                       const float* __restrict__ bias,          // [4H]
                       float* __restrict__ c_state,             // [16, H]
                       unsigned short* __restrict__ h_out,      // [16, H] bf16
                       unsigned short* __restrict__ seq_out,    // optional
                       int seq_stride) {
    __shared__ unsigned short sA[2][BATCH * KDIM];   // 2 x 32 KB = 64 KB

    const int tid = threadIdx.x;
#if CDNA5_HAS_TDM
    if ((tid >> 5) == 0) {            // wave 0 drives the TDM (EXEC uniform)
        unsigned int lds0 = (unsigned int)(uintptr_t)&sA[0][0];
        unsigned int lds1 = (unsigned int)(uintptr_t)&sA[1][0];
        tdm_load_tile_2d(lds0, A1, BATCH, KDIM, (unsigned long long)lda1);
        tdm_load_tile_2d(lds1, A2, BATCH, KDIM, (unsigned long long)lda2);
        __builtin_amdgcn_s_wait_tensorcnt(0);
    }
#else
    // Cooperative staging fallback: 2048 uint4 per matrix, 128 threads
    for (int i = tid; i < BATCH * KDIM / 8; i += 128) {
        int row = i >> 7;                 // 128 uint4 per 1024-elt row
        int c8 = (i & 127) * 8;
        *(uint4*)&sA[0][row * KDIM + c8] = *(const uint4*)&A1[(size_t)row * lda1 + c8];
        *(uint4*)&sA[1][row * KDIM + c8] = *(const uint4*)&A2[(size_t)row * lda2 + c8];
    }
#endif
    __syncthreads();

    const int lane  = tid & 31;
    const int hi    = (lane >= 16);
    const int l16   = lane & 15;
    const int wave  = blockIdx.x * (blockDim.x >> 5) + (tid >> 5);
    const int ncol  = wave * 16;          // column slice base within H
    const int nlane = ncol + l16;         // this lane's column (B/C/D layout)

    // Init the 4 gate accumulators with the bias (column-broadcast).
    v8f acc[4];
#pragma unroll
    for (int g = 0; g < 4; ++g) {
        float bv = bias[g * HDIM + nlane];
#pragma unroll
        for (int r = 0; r < 8; ++r) acc[g][r] = bv;
    }

    // A k order used here (not the ISA table order; B matches): row = lane&15; K = [0..15] for lanes
    // 0-15 and [16..31] for lanes 16-31, contiguous per lane.
    const int aoff0 = l16 * KDIM + (hi ? 16 : 0);
    const int aoff1 = l16 * KDIM + (hi ? 24 : 8);
    // B k order used here (same as A): lane column = lane&15; K = kc+[0..15] (lanes 0-15)
    // or kc+[16..31] (lanes 16-31), contiguous per lane.
    const int bko = hi ? 16 : 0;

    for (int src = 0; src < 2; ++src) {
        const unsigned short* sAp = sA[src];
        const unsigned short* Wp  = src ? W2 : W1;
        for (int kc = 0; kc < KDIM; kc += 32) {
            BF16Frag a;
            a.u[0] = *(const uint4*)&sAp[aoff0 + kc];
            a.u[1] = *(const uint4*)&sAp[aoff1 + kc];
#pragma unroll
            for (int g = 0; g < 4; ++g) {
                BF16Frag b;
                const unsigned short* wrow =
                    Wp + (size_t)(g * HDIM + nlane) * KDIM + kc + bko;
                b.u[0] = *(const uint4*)(wrow);
                b.u[1] = *(const uint4*)(wrow + 8);
                acc[g] = wmma_bf16(a, b, acc[g]);
            }
        }
    }

    // sLSTM cell update directly on the accumulator layout:
    // element r of C tile -> (batch = r + 8*hi, col = nlane)
#pragma unroll
    for (int r = 0; r < 8; ++r) {
        int b = r + (hi ? 8 : 0);
        int idx = b * HDIM + nlane;
        float ig = acc[0][r], fg = acc[1][r], zg = acc[2][r], og = acc[3][r];
        float cprev = c_state[idx];
        float cn = (1.0f / (1.0f + __expf(-fg))) * cprev + __expf(ig) * tanhf(zg);
        float hn = (1.0f / (1.0f + __expf(-og))) * tanhf(cn);
        c_state[idx] = cn;
        unsigned short hb = f32_to_bf16(hn);
        h_out[idx] = hb;
        if (seq_out) seq_out[(size_t)b * seq_stride + nlane] = hb;
    }
}

// ---------------------------------------------------------------------------
// Register-blocked bf16 WMMA GEMM:
//   C[M,N] f32 = A[M,K] bf16 @ W[N,K]^T bf16 + bias[N]
// One wave computes a 32(M) x 64(N) macro-tile = 2x4 WMMA tiles, so each A
// fragment is reused 4x and each B fragment 2x (12 b128 loads per 8 WMMAs).
// Requires M % 32 == 0 and N % 64 == 0 (true for all uses here).
// ---------------------------------------------------------------------------
__global__ __launch_bounds__(256)
void gemm_bf16_kernel(const unsigned short* __restrict__ A,
                      const unsigned short* __restrict__ W,
                      const float* __restrict__ bias,
                      float* __restrict__ C, int M, int N, int K) {
    const int lane = threadIdx.x & 31;
    const int wave = blockIdx.x * (blockDim.x >> 5) + (threadIdx.x >> 5);
    const int ngrp = N >> 6;          // macro-tiles along N
    const int mg = wave / ngrp;
    const int ng = wave % ngrp;
    if (mg >= (M >> 5)) return;       // uniform per wave: EXEC stays all-ones

    const int hi  = (lane >= 16);
    const int l16 = lane & 15;

    v8f acc[2][4] = {};
    const size_t a0 = ((size_t)(mg * 32) + l16) * K;
    const size_t a1 = a0 + (size_t)16 * K;
    const size_t b0 = ((size_t)(ng * 64) + l16) * K;
    const int ao0 = hi ? 16 : 0, ao1 = hi ? 24 : 8;
    const int bo  = hi ? 16 : 0;

    for (int kc = 0; kc < K; kc += 32) {
        BF16Frag a[2], b[4];
        a[0].u[0] = *(const uint4*)&A[a0 + kc + ao0];
        a[0].u[1] = *(const uint4*)&A[a0 + kc + ao1];
        a[1].u[0] = *(const uint4*)&A[a1 + kc + ao0];
        a[1].u[1] = *(const uint4*)&A[a1 + kc + ao1];
#pragma unroll
        for (int j = 0; j < 4; ++j) {
            const unsigned short* wp = &W[b0 + (size_t)j * 16 * K + kc + bo];
            b[j].u[0] = *(const uint4*)(wp);
            b[j].u[1] = *(const uint4*)(wp + 8);
        }
#pragma unroll
        for (int i = 0; i < 2; ++i)
#pragma unroll
            for (int j = 0; j < 4; ++j)
                acc[i][j] = wmma_bf16(a[i], b[j], acc[i][j]);
    }

#pragma unroll
    for (int i = 0; i < 2; ++i) {
#pragma unroll
        for (int j = 0; j < 4; ++j) {
            int col = ng * 64 + j * 16 + l16;
            float bv = bias ? bias[col] : 0.0f;
#pragma unroll
            for (int r = 0; r < 8; ++r) {
                int m = mg * 32 + i * 16 + r + (hi ? 8 : 0);
                C[(size_t)m * N + col] = acc[i][j][r] + bv;
            }
        }
    }
}

// ---------------------------------------------------------------------------
// GELU(exact) + residual + LayerNorm over H=1024, one block per token.
// Reads ypre (proj output, bias already added) and xf (residual); writes the
// new block activations back to xf (fp32) and xb (bf16 GEMM operand).
// ---------------------------------------------------------------------------
__global__ __launch_bounds__(256)
void gelu_res_ln_kernel(const float* __restrict__ ypre,
                        float* __restrict__ xf,
                        unsigned short* __restrict__ xb,
                        const float* __restrict__ g,
                        const float* __restrict__ b) {
    __shared__ float ss[256], sq2[256];
    const int tid = threadIdx.x;
    const size_t base = (size_t)blockIdx.x * HDIM;

    float v[4];
    float s = 0.0f, sq = 0.0f;
#pragma unroll
    for (int j = 0; j < 4; ++j) {
        int e = tid + j * 256;
        float y = ypre[base + e];
        float ge = 0.5f * y * (1.0f + erff(y * 0.70710678f));
        float r = ge + xf[base + e];
        v[j] = r;
        s += r;
        sq += r * r;
    }
    ss[tid] = s;
    sq2[tid] = sq;
    __syncthreads();
    for (int st = 128; st; st >>= 1) {
        if (tid < st) { ss[tid] += ss[tid + st]; sq2[tid] += sq2[tid + st]; }
        __syncthreads();
    }
    float mean = ss[0] * (1.0f / HDIM);
    float var  = sq2[0] * (1.0f / HDIM) - mean * mean;
    float rstd = rsqrtf(var + 1e-5f);
#pragma unroll
    for (int j = 0; j < 4; ++j) {
        int e = tid + j * 256;
        float o = (v[j] - mean) * rstd * g[e] + b[e];
        xf[base + e] = o;
        xb[base + e] = f32_to_bf16(o);
    }
}

// ---------------------------------------------------------------------------
// Host orchestration
// ---------------------------------------------------------------------------
extern "C" void kernel_launch(void* const* d_in, const int* in_sizes, int n_in,
                              void* d_out, int out_size, void* d_ws, size_t ws_size,
                              hipStream_t stream) {
    const int*   seq = (const int*)d_in[0];
    const float* emb = (const float*)d_in[1];
    const float* Wih = (const float*)d_in[2];   // [2,2,4096,1024]
    const float* Whh = (const float*)d_in[3];   // [2,2,4096,1024]
    const float* bg  = (const float*)d_in[4];   // [2,2,4096]
    const float* pW  = (const float*)d_in[5];   // [2,1024,1024]
    const float* pb  = (const float*)d_in[6];   // [2,1024]
    const float* lng = (const float*)d_in[7];   // [2,1024]
    const float* lnb = (const float*)d_in[8];   // [2,1024]
    const float* oW  = (const float*)d_in[9];   // [32000,1024]
    const float* ob  = (const float*)d_in[10];  // [32000]
    float* out = (float*)d_out;

    const size_t nWih = (size_t)2 * 2 * 4096 * 1024;
    const size_t nWhh = nWih;
    const size_t nPw  = (size_t)2 * 1024 * 1024;
    const size_t nOw  = (size_t)32000 * 1024;
    const size_t nTok = (size_t)BATCH * SEQ;            // 4096 tokens
    const size_t nAct = nTok * HDIM;                    // 4.19M elements

    char* ws = (char*)d_ws;
    size_t off = 0;
    auto take = [&](size_t bytes) -> char* {
        char* p = ws + off;
        off += (bytes + 255) & ~(size_t)255;
        return p;
    };

    unsigned short* wihb = (unsigned short*)take(nWih * 2);
    unsigned short* whhb = (unsigned short*)take(nWhh * 2);
    unsigned short* pwb  = (unsigned short*)take(nPw * 2);
    unsigned short* owb  = (unsigned short*)take(nOw * 2);
    float*          xf   = (float*)take(nAct * 4);          // residual path
    unsigned short* xb   = (unsigned short*)take(nAct * 2); // GEMM operand
    unsigned short* hseq = (unsigned short*)take(nAct * 2); // layer-1 outputs
    float*          ypre = (float*)take(nAct * 4);          // proj pre-act
    unsigned short* h0p[2], *h1p[2];
    h0p[0] = (unsigned short*)take(BATCH * HDIM * 2);
    h0p[1] = (unsigned short*)take(BATCH * HDIM * 2);
    h1p[0] = (unsigned short*)take(BATCH * HDIM * 2);
    h1p[1] = (unsigned short*)take(BATCH * HDIM * 2);
    float* c0 = (float*)take(BATCH * HDIM * 4);
    float* c1 = (float*)take(BATCH * HDIM * 4);

    // ---- weights -> bf16 (one-time per launch; ~260 MB read, HBM-cheap) ----
    cvt_bf16_kernel<<<2048, 256, 0, stream>>>(Wih, wihb, nWih);
    cvt_bf16_kernel<<<2048, 256, 0, stream>>>(Whh, whhb, nWhh);
    cvt_bf16_kernel<<<512, 256, 0, stream>>>(pW, pwb, nPw);
    cvt_bf16_kernel<<<4096, 256, 0, stream>>>(oW, owb, nOw);

    // ---- embedding gather ----
    embed_kernel<<<(int)(nAct / 256), 256, 0, stream>>>(seq, emb, xf, xb, (int)nAct);

    for (int blk = 0; blk < 2; ++blk) {
        hipMemsetAsync(h0p[0], 0, BATCH * HDIM * 2, stream);
        hipMemsetAsync(h1p[0], 0, BATCH * HDIM * 2, stream);
        hipMemsetAsync(c0, 0, BATCH * HDIM * 4, stream);
        hipMemsetAsync(c1, 0, BATCH * HDIM * 4, stream);

        const unsigned short* wih0 = wihb + (size_t)(blk * 2 + 0) * 4096 * 1024;
        const unsigned short* wih1 = wihb + (size_t)(blk * 2 + 1) * 4096 * 1024;
        const unsigned short* whh0 = whhb + (size_t)(blk * 2 + 0) * 4096 * 1024;
        const unsigned short* whh1 = whhb + (size_t)(blk * 2 + 1) * 4096 * 1024;
        const float* bg0 = bg + (size_t)(blk * 2 + 0) * 4096;
        const float* bg1 = bg + (size_t)(blk * 2 + 1) * 4096;

        for (int t = 0; t < SEQ; ++t) {
            int pin = t & 1, pout = pin ^ 1;   // double-buffered hidden state
            // layer 0: gates = x_t @ Wih0^T + h0 @ Whh0^T + bg0
            slstm_cell_kernel<<<16, 128, 0, stream>>>(
                xb + (size_t)t * HDIM, SEQ * HDIM,   // x rows: stride S*H
                wih0,
                h0p[pin], HDIM, whh0,
                bg0, c0, h0p[pout],
                nullptr, 0);
            // layer 1: gates = h0_new @ Wih1^T + h1 @ Whh1^T + bg1; emit h1_t
            slstm_cell_kernel<<<16, 128, 0, stream>>>(
                h0p[pout], HDIM, wih1,
                h1p[pin], HDIM, whh1,
                bg1, c1, h1p[pout],
                hseq + (size_t)t * HDIM, SEQ * HDIM);
        }

        // projection: ypre = hseq @ pW^T + pb   (M=4096, N=1024, K=1024)
        {
            int waves = (4096 / 32) * (1024 / 64);
            gemm_bf16_kernel<<<(waves + 7) / 8, 256, 0, stream>>>(
                hseq, pwb + (size_t)blk * 1024 * 1024, pb + (size_t)blk * 1024,
                ypre, 4096, 1024, 1024);
        }
        // gelu + residual + layernorm -> new xf / xb
        gelu_res_ln_kernel<<<(int)nTok, 256, 0, stream>>>(
            ypre, xf, xb, lng + (size_t)blk * 1024, lnb + (size_t)blk * 1024);
    }

    // logits = x @ out_W^T + out_b   (M=4096, N=32000, K=1024)
    {
        int waves = (4096 / 32) * (32000 / 64);
        gemm_bf16_kernel<<<(waves + 7) / 8, 256, 0, stream>>>(
            xb, owb, ob, out, 4096, 32000, 1024);
    }
}
